// SequentialLora_59459527246486
// MI455X (gfx1250) — matrix operand order repaired and checked by reading, not compiled, not run on hardware
//
#include <hip/hip_runtime.h>

typedef __attribute__((ext_vector_type(16))) _Float16 v16h;
typedef __attribute__((ext_vector_type(8)))  _Float16 v8h;
typedef __attribute__((ext_vector_type(8)))  float    v8f;

#define DIN  4096
#define DOUT 4096
#define HMAX 64   // h stride per token in workspace (max rank)

// ---------------------------------------------------------------------------
// Stage 1: h[token][0..R) = x[token] . A[wid]   (f32 accumulate via WMMA)
// Grid: 64 tokens * NT column-tiles; block = 256 threads = 8 waves that split
// the K=4096 reduction (16 WMMAs each), then a 1KB LDS cross-wave reduce.
// WMMA trick: A-operand = x chunk replicated across all 16 M rows, so D row 0
// is the GEMV result while memory traffic stays identical to the ideal GEMV.
// ---------------------------------------------------------------------------
template<int R, int NT>
__global__ void __launch_bounds__(256)
lora_stage1(const _Float16* __restrict__ x,
            const int* __restrict__ wids,
            const _Float16* __restrict__ Aall,
            float* __restrict__ h_ws,
            int token_base)
{
    constexpr int NCOLS = (R < 16) ? R : 16;     // valid columns in this tile
    __shared__ float red[8][32];

    const int local = blockIdx.x / NT;           // token within group (NT is 2^k)
    const int tile  = blockIdx.x % NT;
    const int token = token_base + local;
    const int wid   = wids[local];
    const _Float16* __restrict__ A =
        Aall + (size_t)wid * DIN * R + tile * 16;         // column tile base
    const _Float16* __restrict__ xp = x + (size_t)token * DIN;

    const int lane  = threadIdx.x & 31;
    const int wave  = threadIdx.x >> 5;
    const int half8 = (lane & 16) ? 8 : 0;       // A-operand K offset per half-wave

    v8f acc = {};
    const int k_begin = wave * (DIN / 8);        // 512-wide K slice per wave
#pragma unroll 2
    for (int k0 = k_begin; k0 < k_begin + (DIN / 8); k0 += 32) {
        // A-operand: x[k0..k0+31] replicated over M (ISA 16-bit A 16x32 layout)
        v8h p0 = *(const v8h*)(xp + k0 + half8);
        v8h p1 = *(const v8h*)(xp + k0 + 16 + half8);
        v16h a;
#pragma unroll
        for (int e = 0; e < 8; ++e) { a[e] = p0[e]; a[8 + e] = p1[e]; }

        // B-operand: column N = lane%16 (zero if >= NCOLS), K in table order matching a
        v16h b;
        {
            const int col = lane & 15;
#pragma unroll
            for (int e = 0; e < 16; ++e) {
                const int kk = (e < 8) ? (half8 + e) : (16 + half8 + (e - 8));
                b[e] = (col < NCOLS) ? A[(size_t)(k0 + kk) * R + col] : (_Float16)0.0f;
            }
        }

        acc = __builtin_amdgcn_wmma_f32_16x16x32_f16(false, a, false, b,
                                                     (short)0, acc, false, false);
    }

    // Cross-wave K reduction of D row 0 (all M rows identical by construction)
    red[wave][lane] = acc[0];
    __syncthreads();
    if (wave == 0 && lane < NCOLS) {
        float s = 0.f;
#pragma unroll
        for (int w = 0; w < 8; ++w) s += red[w][lane];
        h_ws[token * HMAX + tile * 16 + lane] = s;
    }
}

// ---------------------------------------------------------------------------
// Stage 2: y[token] = 2 * (h[token] . B[wid]).  One block per token; each of
// 8 waves owns 32 output tiles of 16.  A-operand = h replicated (built once);
// per tile: per lane up to 16 DOUT-strided 2B B loads per K chunk + 1-2 WMMAs + b16 stores.
// ---------------------------------------------------------------------------
template<int R>
__global__ void __launch_bounds__(256)
lora_stage2(const int* __restrict__ wids,
            const _Float16* __restrict__ Ball,
            const float* __restrict__ h_ws,
            _Float16* __restrict__ out,
            int token_base)
{
    const int local = blockIdx.x;
    const int token = token_base + local;
    const int wid   = wids[local];
    const _Float16* __restrict__ B = Ball + (size_t)wid * R * DOUT;
    const float*    __restrict__ h = h_ws + token * HMAX;

    const int lane  = threadIdx.x & 31;
    const int wave  = threadIdx.x >> 5;
    const int half8 = (lane & 16) ? 8 : 0;

    // Replicated-h A-operand chunk(s); K dim = R (two K=32 chunks iff R==64)
    v16h a0, a1;
#pragma unroll
    for (int e = 0; e < 16; ++e) {
        const int kk = (e < 8) ? (half8 + e) : (16 + half8 + (e - 8));
        a0[e] = (kk < R) ? (_Float16)h[kk] : (_Float16)0.0f;
        if constexpr (R == 64) a1[e] = (_Float16)h[32 + kk];
        else                   a1[e] = (_Float16)0.0f;
    }

    const int tile_lo = wave * 32;
#pragma unroll 2
    for (int tile = tile_lo; tile < tile_lo + 32; ++tile) {
        const int o0 = tile * 16;
        v8f acc = {};

        // K chunk 0: column N = o0 + lane%16, K in table order matching a0 (zero if K >= R)
        v16h b = {};
#pragma unroll
        for (int e = 0; e < 16; ++e) {
            const int kk = (e < 8) ? (half8 + e) : (16 + half8 + (e - 8));
            b[e] = (kk < R) ? B[(size_t)kk * DOUT + o0 + (lane & 15)] : (_Float16)0.0f;
        }
        acc = __builtin_amdgcn_wmma_f32_16x16x32_f16(false, a0, false, b,
                                                     (short)0, acc, false, false);
        if constexpr (R == 64) {   // K chunk 1: K = 32 + table order, column N = o0 + lane%16
            v16h b2;
#pragma unroll
            for (int e = 0; e < 16; ++e) {
                const int kk = (e < 8) ? (half8 + e) : (16 + half8 + (e - 8));
                b2[e] = B[(size_t)(32 + kk) * DOUT + o0 + (lane & 15)];
            }
            acc = __builtin_amdgcn_wmma_f32_16x16x32_f16(false, a1, false, b2,
                                                         (short)0, acc, false, false);
        }

        // D row 0 (lanes 0..15, VGPR0) holds y[o0+lane]; scale by 2, store fp16
        if (lane < 16) {
            out[(size_t)token * DOUT + o0 + lane] = (_Float16)(2.0f * acc[0]);
        }
    }
}

// ---------------------------------------------------------------------------
extern "C" void kernel_launch(void* const* d_in, const int* in_sizes, int n_in,
                              void* d_out, int out_size, void* d_ws, size_t ws_size,
                              hipStream_t stream)
{
    (void)in_sizes; (void)n_in; (void)out_size; (void)ws_size;
    // setup_inputs() dict order: x, (wids_0, A_0, B_0), (wids_1, A_1, B_1), ...
    const _Float16* x  = (const _Float16*)d_in[0];
    const int*      w0 = (const int*)d_in[1];
    const _Float16* A0 = (const _Float16*)d_in[2];
    const _Float16* B0 = (const _Float16*)d_in[3];
    const int*      w1 = (const int*)d_in[4];
    const _Float16* A1 = (const _Float16*)d_in[5];
    const _Float16* B1 = (const _Float16*)d_in[6];
    const int*      w2 = (const int*)d_in[7];
    const _Float16* A2 = (const _Float16*)d_in[8];
    const _Float16* B2 = (const _Float16*)d_in[9];
    const int*      w3 = (const int*)d_in[10];
    const _Float16* A3 = (const _Float16*)d_in[11];
    const _Float16* B3 = (const _Float16*)d_in[12];

    float*    h_ws = (float*)d_ws;            // 256 tokens * 64 floats = 64 KB
    _Float16* out  = (_Float16*)d_out;

    // Stage 1 (h = x.A), specialized per rank group
    lora_stage1<8,  1><<< 64, 256, 0, stream>>>(x, w0, A0, h_ws, 0);
    lora_stage1<16, 1><<< 64, 256, 0, stream>>>(x, w1, A1, h_ws, 64);
    lora_stage1<32, 2><<<128, 256, 0, stream>>>(x, w2, A2, h_ws, 128);
    lora_stage1<64, 4><<<256, 256, 0, stream>>>(x, w3, A3, h_ws, 192);

    // Stage 2 (y = 2 * h.B), specialized per rank group
    lora_stage2<8 ><<<64, 256, 0, stream>>>(w0, B0, h_ws, out, 0);
    lora_stage2<16><<<64, 256, 0, stream>>>(w1, B1, h_ws, out, 64);
    lora_stage2<32><<<64, 256, 0, stream>>>(w2, B2, h_ws, out, 128);
    lora_stage2<64><<<64, 256, 0, stream>>>(w3, B3, h_ws, out, 192);
}
